// ChatGLM2Attention_49014166782353
// MI455X (gfx1250) — matrix operand order repaired and checked by reading, not compiled, not run on hardware
//
#include <hip/hip_runtime.h>
#include <cstddef>
#include <cstdint>

// ---------------- CDNA5 WMMA types ----------------
typedef __bf16 v16bf __attribute__((ext_vector_type(16)));
typedef __bf16 v8bf  __attribute__((ext_vector_type(8)));
typedef float  v8f   __attribute__((ext_vector_type(8)));
typedef unsigned int u32x4 __attribute__((ext_vector_type(4)));
typedef unsigned int u32x8 __attribute__((ext_vector_type(8)));

union bf16x16 { v16bf v; v8bf h[2]; };

__device__ __forceinline__ v8f wmma_bf16(v16bf a, v16bf b, v8f c) {
    // D = A(16x32 bf16) * B(32x16 bf16) + C(16x16 f32)
    return __builtin_amdgcn_wmma_f32_16x16x32_bf16(
        /*neg_a=*/false, a, /*neg_b=*/false, b,
        /*c_mod=*/(short)0, c, /*reuse_a=*/false, /*reuse_b=*/false);
}

__device__ __forceinline__ unsigned short f32_to_bf16(float f) {
    union { float f; unsigned u; } v; v.f = f;
    unsigned r = v.u + 0x7FFFu + ((v.u >> 16) & 1u);   // RNE
    return (unsigned short)(r >> 16);
}

// ---- CDNA5 async / TDM helpers (inline asm; portable across toolchains) ----
__device__ __forceinline__ void async_load_b128(unsigned lds_addr, const void* gaddr) {
    // GLOBAL_LOAD_ASYNC_TO_LDS_B128, GV mode (SADDR=NULL). VDST = LDS byte addr.
    asm volatile("global_load_async_to_lds_b128 %0, %1, off"
                 :: "v"(lds_addr), "v"(gaddr) : "memory");
}
__device__ __forceinline__ void wait_async0() {
    asm volatile("s_wait_asynccnt 0x0" ::: "memory");
}
__device__ __forceinline__ void tdm_load_2d(u32x4 g0, u32x8 g1) {
    // TENSOR_LOAD_TO_LDS, 2-group (<=2D) form: VADDR2/3 = NULL
    asm volatile("tensor_load_to_lds %0, %1" :: "s"(g0), "s"(g1) : "memory");
}
__device__ __forceinline__ void wait_tensor0() {
#if __has_builtin(__builtin_amdgcn_s_wait_tensorcnt)
    __builtin_amdgcn_s_wait_tensorcnt(0);
#else
    asm volatile("s_wait_tensorcnt 0x0" ::: "memory");
#endif
}
__device__ __forceinline__ unsigned lds_off(const void* p) {
    // generic LDS pointer: low 32 bits are the LDS byte address (aperture rule)
    return (unsigned)(uintptr_t)p;
}

// ---------------- problem constants ----------------
#define BB      2
#define SS      2048
#define NSTATE  4096
#define NHEAD   32
#define DHEAD   128
#define NGROUP  2
#define HG      16                  // heads per group
#define QKVN    4608                // DHEAD*(NHEAD+2*NGROUP)
#define MROWS   (BB*SS)             // 4096

// ---------------- 1) f32 -> bf16 convert ----------------
__global__ void cvt_f32_bf16(const float* __restrict__ src,
                             unsigned short* __restrict__ dst, int n) {
    int i = blockIdx.x * blockDim.x + threadIdx.x;
    int stride = gridDim.x * blockDim.x;
    for (; i < n; i += stride) dst[i] = f32_to_bf16(src[i]);
}

// ---------------- 2/5) bf16 WMMA GEMM: C[M][N] = A[M][K] * Wt[N][K]^T + bias ----------------
// One wave computes a 16(M) x 64(N) tile (4 C fragments), K-loop in steps of 32.
// Block = 256 threads = 8 waves -> 16 x 512 tile per block.
__global__ __launch_bounds__(256) void gemm_bf16_wmma(
    const unsigned short* __restrict__ A,   // [M][K] bf16 bits
    const unsigned short* __restrict__ Wt,  // [N][K] bf16 bits (row-major "B^T")
    const float* __restrict__ bias,         // [N] or nullptr
    float* __restrict__ C,                  // [M][N] f32
    int M, int N, int K)
{
    const int wave = threadIdx.x >> 5;
    const int lane = threadIdx.x & 31;
    const int n16  = lane & 15;
    const int half = lane >> 4;
    const int m0 = blockIdx.y * 16;
    const int n0 = blockIdx.x * 512 + wave * 64;

    v8f c[4];
#pragma unroll
    for (int j = 0; j < 4; ++j) {
        float bv = bias ? bias[n0 + 16 * j + n16] : 0.0f;
#pragma unroll
        for (int r = 0; r < 8; ++r) c[j][r] = bv;
    }

    const unsigned short* arow = A + (size_t)(m0 + n16) * K + half * 16;
    const unsigned short* brow0 = Wt + (size_t)(n0 + n16) * K + half * 16;

    for (int k = 0; k < K; k += 32) {
        __builtin_prefetch(arow + k + 256, 0, 1);          // global_prefetch_b8
        bf16x16 a;
        a.h[0] = *(const v8bf*)(arow + k);                 // K = kb .. kb+7
        a.h[1] = *(const v8bf*)(arow + k + 8);             // K = kb+8 .. kb+15
#pragma unroll
        for (int j = 0; j < 4; ++j) {
            v16bf b = *(const v16bf*)(brow0 + (size_t)(16 * j) * K + k); // 16 contiguous K
            c[j] = wmma_bf16(a.v, b, c[j]);
        }
    }

#pragma unroll
    for (int j = 0; j < 4; ++j)
#pragma unroll
        for (int r = 0; r < 8; ++r) {
            int row = m0 + r + 8 * half;
            C[(size_t)row * N + n0 + 16 * j + n16] = c[j][r];
        }
}

// ---------------- 3) RoPE + scatter ----------------
__global__ __launch_bounds__(256) void rope_scatter(
    const float* __restrict__ fused, const float* __restrict__ freqs,
    float* __restrict__ k_out, float* __restrict__ v_out,
    unsigned short* __restrict__ qbf, unsigned short* __restrict__ kbf,
    unsigned short* __restrict__ vbf)
{
    const int bs = blockIdx.x;                   // 0 .. B*S-1
    const int b = bs / SS, s = bs % SS;
    const float* frow = fused + (size_t)bs * QKVN;
    const float* fc   = freqs + (size_t)bs * (DHEAD);   // [64][2]
    const int t = threadIdx.x;
    const float scale = 0.08838834764831845f;    // 1/sqrt(128)

    // q: 2048 (cos,sin) pairs, roped + pre-scaled
    for (int i = t; i < 2048; i += 256) {
        int g = i >> 10, h = (i >> 6) & 15, j = i & 63;
        float xr = frow[2 * i], xi = frow[2 * i + 1];
        float cs = fc[2 * j], sn = fc[2 * j + 1];
        float orr = xr * cs - xi * sn;
        float oii = xr * sn + xi * cs;
        size_t qi = ((((size_t)b * NGROUP + g) * HG + h) * SS + s) * DHEAD + 2 * j;
        qbf[qi]     = f32_to_bf16(orr * scale);
        qbf[qi + 1] = f32_to_bf16(oii * scale);
    }
    // k: 128 pairs
    if (t < 128) {
        int g = t >> 6, j = t & 63;
        float xr = frow[NSTATE + 2 * t], xi = frow[NSTATE + 2 * t + 1];
        float cs = fc[2 * j], sn = fc[2 * j + 1];
        float orr = xr * cs - xi * sn;
        float oii = xr * sn + xi * cs;
        size_t ko = (size_t)bs * (NGROUP * DHEAD) + g * DHEAD + 2 * j;
        k_out[ko] = orr; k_out[ko + 1] = oii;
        size_t ki = (((size_t)b * NGROUP + g) * SS + s) * DHEAD + 2 * j;
        kbf[ki]     = f32_to_bf16(orr);
        kbf[ki + 1] = f32_to_bf16(oii);
    }
    // v: faithful overlapping slice fused[..., 256:512]
    if (t < 256) {
        int g = t >> 7, d = t & 127;
        float val = frow[NGROUP * DHEAD + t];
        v_out[(size_t)bs * (NGROUP * DHEAD) + t] = val;
        vbf[(((size_t)b * NGROUP + g) * SS + s) * DHEAD + d] = f32_to_bf16(val);
    }
}

// ---------------- 4) flash attention (bf16 WMMA + TDM/async staging) ----------------
// block = 128 threads = 4 waves, each wave a 16-query tile; waves share the
// 32-key K block (TDM-loaded) / V^T block staged in LDS.
__global__ __launch_bounds__(128) void attn_kernel(
    const unsigned short* __restrict__ qbf,  // [B][G][HG][S][D]
    const unsigned short* __restrict__ kbf,  // [B][G][S][D]
    const unsigned short* __restrict__ vbf,  // [B][G][S][D]
    const float* __restrict__ mask,          // [B][S][S]
    unsigned short* __restrict__ attnbf)     // [B*S][NHEAD*DHEAD]
{
    __shared__ __align__(32) unsigned short Kbuf[32 * DHEAD];   // [key][dim]
    __shared__ __align__(32) unsigned short VbufT[DHEAD * 32];  // [dim][key]
    __shared__ __align__(16) unsigned short Pbuf[4][16 * 32];   // per-wave P tile
    __shared__ __align__(16) unsigned short Qbuf[4][16 * DHEAD];// per-wave Q tile

    const int head = blockIdx.y;            // 0 .. B*NHEAD-1
    const int b = head >> 5;
    const int g = (head & 31) >> 4;
    const int h = head & 15;
    const int wave = threadIdx.x >> 5;
    const int lane = threadIdx.x & 31;
    const int n16 = lane & 15;
    const int half = lane >> 4;
    const int m0 = blockIdx.x * 64 + wave * 16;

    const unsigned short* qp = qbf + ((((size_t)b * NGROUP + g) * HG + h) * SS) * DHEAD;
    const unsigned short* kp = kbf + (((size_t)b * NGROUP + g) * SS) * DHEAD;
    const unsigned short* vp = vbf + (((size_t)b * NGROUP + g) * SS) * DHEAD;

    // --- async preload of this wave's 16x128 Q tile into LDS (no VGPR roundtrip) ---
    {
        unsigned qlbase = lds_off(&Qbuf[wave][0]);
        for (int i = lane; i < 256; i += 32) {          // 256 x 16B chunks
            int row = i >> 4, c8 = (i & 15) * 8;
            async_load_b128(qlbase + i * 16, qp + (size_t)(m0 + row) * DHEAD + c8);
        }
        wait_async0();                                  // own-wave data ready
    }
    // read Q A-fragments from LDS
    bf16x16 qa[4];
#pragma unroll
    for (int f = 0; f < 4; ++f) {
        qa[f].h[0] = *(const v8bf*)(&Qbuf[wave][n16 * DHEAD + 32 * f + half * 16]);
        qa[f].h[1] = *(const v8bf*)(&Qbuf[wave][n16 * DHEAD + 32 * f + half * 16 + 8]);
    }

    // --- TDM descriptor group1 (k0-invariant): bf16 tile 128(dim) x 32(keys) ---
    u32x8 tg1;
    tg1[0] = 0x00010000u;          // workgroup_mask=0, data_size=1 (2 bytes)
    tg1[1] = (unsigned)(DHEAD) << 16;   // tensor_dim0 = 128 (bits 79:48, low half)
    tg1[2] = 32u << 16;            // tensor_dim0 hi=0 | tensor_dim1 = 32 (low half)
    tg1[3] = (unsigned)(DHEAD) << 16;   // tensor_dim1 hi=0 | tile_dim0 = 128
    tg1[4] = 32u;                  // tile_dim1 = 32, tile_dim2 = 0
    tg1[5] = (unsigned)(DHEAD);    // tensor_dim0_stride = 128 (low 32)
    tg1[6] = 0u;                   // stride0 hi | stride1 lo
    tg1[7] = 0u;                   // stride1 hi (unused, 2D tile)
    const unsigned klds = lds_off(&Kbuf[0]);

    v8f acc[8];
    float mrun[8], lrun[8];
#pragma unroll
    for (int c = 0; c < 8; ++c) {
#pragma unroll
        for (int r = 0; r < 8; ++r) acc[c][r] = 0.0f;
    }
#pragma unroll
    for (int r = 0; r < 8; ++r) { mrun[r] = -3.0e38f; lrun[r] = 0.0f; }

    for (int k0 = 0; k0 < SS; k0 += 32) {
        __syncthreads();                       // previous block fully consumed
        if (wave == 0) {
            // --- Tensor Data Mover: K block global -> LDS, one instruction ---
            uint64_t ga = (uint64_t)(uintptr_t)(kp + (size_t)k0 * DHEAD);
            u32x4 tg0;
            tg0[0] = 1u;                               // count=1, user descriptor
            tg0[1] = klds;                             // lds_addr
            tg0[2] = (unsigned)ga;                     // global_addr low
            tg0[3] = (unsigned)((ga >> 32) & 0x1FFFFFFu) | (2u << 30); // addr hi | type=2
            tdm_load_2d(tg0, tg1);
            wait_tensor0();
        }
        {   // cooperative V^T staging (transpose: TDM/async cannot do this)
            int key = threadIdx.x >> 2;
            int seg = (threadIdx.x & 3) * 32;
            const unsigned short* vsrc = vp + (size_t)(k0 + key) * DHEAD + seg;
#pragma unroll
            for (int i = 0; i < 32; ++i) VbufT[(seg + i) * 32 + key] = vsrc[i];
        }
        __syncthreads();

        // S = Q K^T for 32 keys: two 16x16 tiles, K-dim = 128 (4 WMMAs each)
        v8f st[2];
#pragma unroll
        for (int t = 0; t < 2; ++t) {
#pragma unroll
            for (int r = 0; r < 8; ++r) st[t][r] = 0.0f;
#pragma unroll
            for (int f = 0; f < 4; ++f) {
                v16bf kb = *(const v16bf*)(Kbuf + (16 * t + n16) * DHEAD + 32 * f + half * 16);
                st[t] = wmma_bf16(qa[f].v, kb, st[t]);
            }
        }
        // + attention mask (faithful)
        const float* mrow = mask + ((size_t)b * SS + m0) * SS + k0;
#pragma unroll
        for (int r = 0; r < 8; ++r) {
            int row = r + 8 * half;
            st[0][r] += mrow[(size_t)row * SS + n16];
            st[1][r] += mrow[(size_t)row * SS + 16 + n16];
        }
        // online softmax per row (rows live in VGPR r, lane-half)
#pragma unroll
        for (int r = 0; r < 8; ++r) {
            float mx = fmaxf(st[0][r], st[1][r]);
            mx = fmaxf(mx, __shfl_xor(mx, 1, 32));
            mx = fmaxf(mx, __shfl_xor(mx, 2, 32));
            mx = fmaxf(mx, __shfl_xor(mx, 4, 32));
            mx = fmaxf(mx, __shfl_xor(mx, 8, 32));
            float nm = fmaxf(mrun[r], mx);
            float corr = __expf(mrun[r] - nm);
            float p0 = __expf(st[0][r] - nm);
            float p1 = __expf(st[1][r] - nm);
            float sum = p0 + p1;
            sum += __shfl_xor(sum, 1, 32);
            sum += __shfl_xor(sum, 2, 32);
            sum += __shfl_xor(sum, 4, 32);
            sum += __shfl_xor(sum, 8, 32);
            lrun[r] = lrun[r] * corr + sum;
            mrun[r] = nm;
#pragma unroll
            for (int c = 0; c < 8; ++c) acc[c][r] *= corr;
            int row = r + 8 * half;
            Pbuf[wave][row * 32 + n16]      = f32_to_bf16(p0);
            Pbuf[wave][row * 32 + 16 + n16] = f32_to_bf16(p1);
        }
        // P tile C-layout -> A-fragment via per-wave LDS scratch (in-order DS)
        bf16x16 pf;
        pf.h[0] = *(const v8bf*)(&Pbuf[wave][n16 * 32 + half * 16]);
        pf.h[1] = *(const v8bf*)(&Pbuf[wave][n16 * 32 + half * 16 + 8]);
        // O += P V : 8 N-chunks of 16 dims
#pragma unroll
        for (int c = 0; c < 8; ++c) {
            v16bf vf = *(const v16bf*)(VbufT + (16 * c + n16) * 32 + half * 16);
            acc[c] = wmma_bf16(pf.v, vf, acc[c]);
        }
    }

    // epilogue: normalize and store bf16 in [b,s, (g*16+h)*128 + d] order
#pragma unroll
    for (int c = 0; c < 8; ++c)
#pragma unroll
        for (int r = 0; r < 8; ++r) {
            int row = m0 + r + 8 * half;
            float o = acc[c][r] / lrun[r];
            attnbf[((size_t)b * SS + row) * NSTATE +
                   ((g * HG + h) * DHEAD + 16 * c + n16)] = f32_to_bf16(o);
        }
}

// ---------------- launch ----------------
extern "C" void kernel_launch(void* const* d_in, const int* in_sizes, int n_in,
                              void* d_out, int out_size, void* d_ws, size_t ws_size,
                              hipStream_t stream) {
    const float* x      = (const float*)d_in[0];  // [B,S,4096]
    const float* freqs  = (const float*)d_in[1];  // [B,S,1,1,64,2]
    const float* mask   = (const float*)d_in[2];  // [B,S,S]
    const float* qkv_w  = (const float*)d_in[3];  // [4608,4096]
    const float* qkv_b  = (const float*)d_in[4];  // [4608]
    const float* o_w    = (const float*)d_in[5];  // [4096,4096]

    float* out   = (float*)d_out;                          // [B,S,4096]
    float* k_out = out + (size_t)MROWS * NSTATE;           // [B,S,G,1,128]
    float* v_out = k_out + (size_t)MROWS * NGROUP * DHEAD; // [B,S,G,1,128]

    // workspace layout (bytes, 256-aligned by construction)
    char* w = (char*)d_ws;
    unsigned short* xbf   = (unsigned short*)(w);                       // 4096*4096*2
    unsigned short* wqbf  = (unsigned short*)(w + 33554432ull);         // 4608*4096*2
    unsigned short* owbf  = (unsigned short*)(w + 71303168ull);         // 4096*4096*2
    float*          fused = (float*)        (w + 104857600ull);         // 4096*4608*4
    unsigned short* qbf   = (unsigned short*)(w + 180355072ull);        // 16M*2
    unsigned short* kbf   = (unsigned short*)(w + 213909504ull);        // 1M*2
    unsigned short* vbf   = (unsigned short*)(w + 216006656ull);        // 1M*2
    unsigned short* attnbf = xbf;   // alias: xbf dead after QKV GEMM

    // 1) bf16 conversions
    cvt_f32_bf16<<<4096, 256, 0, stream>>>(x, xbf, MROWS * NSTATE);
    cvt_f32_bf16<<<4096, 256, 0, stream>>>(qkv_w, wqbf, QKVN * NSTATE);
    cvt_f32_bf16<<<4096, 256, 0, stream>>>(o_w, owbf, NSTATE * NSTATE);

    // 2) fused = x @ qkv_w^T + b   (M=4096, N=4608, K=4096)
    gemm_bf16_wmma<<<dim3(QKVN / 512, MROWS / 16), 256, 0, stream>>>(
        xbf, wqbf, qkv_b, fused, MROWS, QKVN, NSTATE);

    // 3) RoPE + scatter (also writes fp32 k/v outputs)
    rope_scatter<<<MROWS, 256, 0, stream>>>(fused, freqs, k_out, v_out, qbf, kbf, vbf);

    // 4) attention: grid = (S/64 query tiles, B*NHEAD heads)
    attn_kernel<<<dim3(SS / 64, BB * NHEAD), 128, 0, stream>>>(
        qbf, kbf, vbf, mask, attnbf);

    // 5) out = attn @ o_w^T  (M=4096, N=4096, K=4096) -> fp32 d_out
    gemm_bf16_wmma<<<dim3(NSTATE / 512, MROWS / 16), 256, 0, stream>>>(
        attnbf, owbf, nullptr, out, MROWS, NSTATE, NSTATE);
}
